// Pool_55594056680086
// MI455X (gfx1250) — 8-bit matrix operand order repaired and checked by reading, compiled, not run on hardware
//
#include <hip/hip_runtime.h>
#include <hip/hip_bf16.h>

typedef __attribute__((ext_vector_type(8))) int v8i;

#define N_NODES 8192
#define DFEAT   512
#define KSEL    4096

#if defined(__gfx1250__) && __has_builtin(__builtin_amdgcn_global_load_async_to_lds_b128)
#define USE_ASYNC_LDS 1
#else
#define USE_ASYNC_LDS 0
#endif

#if USE_ASYNC_LDS
#if __has_builtin(__builtin_amdgcn_s_wait_asynccnt)
#define WAIT_ASYNC(n) __builtin_amdgcn_s_wait_asynccnt(n)
#else
#define WAIT_ASYNC(n) asm volatile("s_wait_asynccnt %0" ::"i"(n) : "memory")
#endif
typedef int v4i_gcc __attribute__((vector_size(16)));
typedef __attribute__((address_space(1))) void* gvoid_t;
typedef __attribute__((address_space(3))) void* lvoid_t;
typedef __attribute__((address_space(1))) v4i_gcc* gvec_t;
typedef __attribute__((address_space(3))) v4i_gcc* lvec_t;
__device__ __forceinline__ void async_copy32B(const unsigned char* gsrc,
                                              unsigned char* ldst) {
  gvec_t s = (gvec_t)(gvoid_t)gsrc;
  lvec_t d = (lvec_t)(lvoid_t)ldst;
  __builtin_amdgcn_global_load_async_to_lds_b128(s, d, 0, 0);
  __builtin_amdgcn_global_load_async_to_lds_b128(s, d, 16, 0);
}
#else
#define WAIT_ASYNC(n)
#endif

__global__ __launch_bounds__(256) void scores_kernel(
    const float* __restrict__ h, const float* __restrict__ W,
    const float* __restrict__ b, float* __restrict__ scores) {
  int lane = threadIdx.x & 31;
  int wave = threadIdx.x >> 5;
  int row  = blockIdx.x * 8 + wave;
  const float* hr = h + (size_t)row * DFEAT;
  float acc = 0.f;
  for (int k = lane; k < DFEAT; k += 32) acc += hr[k] * W[k];
  for (int off = 16; off > 0; off >>= 1) acc += __shfl_xor(acc, off, 32);
  if (lane == 0) {
    float z = acc + b[0];
    scores[row] = 1.f / (1.f + __expf(-z));
  }
}

__global__ __launch_bounds__(1024) void topk_kernel(
    const float* __restrict__ scores, int* __restrict__ idx_out,
    float* __restrict__ val_out, int* __restrict__ rank,
    float* __restrict__ out_idx_f) {
  __shared__ unsigned long long keys[N_NODES];
  int t = threadIdx.x;
  for (int i = t; i < N_NODES; i += 1024) {
    unsigned sb = __float_as_uint(scores[i]);
    keys[i] = ((unsigned long long)sb << 32) | (unsigned)(0xFFFFFFFFu - (unsigned)i);
    rank[i] = -1;
  }
  __syncthreads();
  for (int k = 2; k <= N_NODES; k <<= 1) {
    for (int j = k >> 1; j > 0; j >>= 1) {
      for (int i = t; i < N_NODES; i += 1024) {
        int p = i ^ j;
        if (p > i) {
          bool desc = ((i & k) == 0);
          unsigned long long a = keys[i], bkey = keys[p];
          bool sw = desc ? (a < bkey) : (a > bkey);
          if (sw) { keys[i] = bkey; keys[p] = a; }
        }
      }
      __syncthreads();
    }
  }
  for (int i = t; i < KSEL; i += 1024) {
    unsigned long long key = keys[i];
    int id  = (int)(0xFFFFFFFFu - (unsigned)(key & 0xFFFFFFFFu));
    float v = __uint_as_float((unsigned)(key >> 32));
    idx_out[i]   = id;
    val_out[i]   = v;
    out_idx_f[i] = (float)id;
    rank[id]     = i;
  }
}

__global__ __launch_bounds__(256) void newh_kernel(
    const float* __restrict__ h, const int* __restrict__ idx,
    const float* __restrict__ vals, float* __restrict__ out_h) {
  int i = blockIdx.x;
  int id = idx[i];
  float v = vals[i];
  const float* src = h + (size_t)id * DFEAT;
  float* dst = out_h + (size_t)i * DFEAT;
  for (int d = threadIdx.x; d < DFEAT; d += blockDim.x) dst[d] = src[d] * v;
}

__global__ __launch_bounds__(256) void pack_kernel(
    const float* __restrict__ g, const int* __restrict__ rank,
    unsigned char* __restrict__ A, unsigned char* __restrict__ Bt) {
  __shared__ unsigned char tileT[64 * 65];
  int bx = blockIdx.x & 127, by = blockIdx.x >> 7;
  int c0 = bx * 64, r0 = by * 64;
  int tx = threadIdx.x & 63, tg = threadIdx.x >> 6;
  for (int it = 0; it < 16; ++it) {
    int rr = tg + it * 4;
    int r = r0 + rr;
    float f = g[(size_t)r * N_NODES + c0 + tx];
    unsigned char v = (f != 0.f) ? 1 : 0;
    int ra = rank[r];
    if (ra >= 0) A[(size_t)ra * N_NODES + c0 + tx] = v;
    tileT[tx * 65 + rr] = v;
  }
  __syncthreads();
  for (int it = 0; it < 16; ++it) {
    int cc = tg + it * 4;
    int rb = rank[c0 + cc];
    if (rb >= 0) Bt[(size_t)rb * N_NODES + r0 + tx] = tileT[cc * 65 + tx];
  }
}

__global__ __launch_bounds__(256) void gemm2hop_kernel(
    const unsigned char* __restrict__ A, const unsigned char* __restrict__ Bt,
    float* __restrict__ out_g) {
  __shared__ unsigned char As[2][128 * 80];
  __shared__ unsigned char Bs[2][128 * 80];
  int i0 = (int)blockIdx.y * 128;
  int j0 = (int)blockIdx.x * 128;
  int t = threadIdx.x;
  int lane = t & 31, w = t >> 5;
  int wr = w & 1;
  int wc = w >> 1;

  v8i acc[4][2];
#pragma unroll
  for (int mi = 0; mi < 4; ++mi)
#pragma unroll
    for (int nj = 0; nj < 2; ++nj) acc[mi][nj] = (v8i){};

  int mrow  = lane & 15;
  int abase = (lane < 16) ? 0 : 8;
  int bbase = (lane < 16) ? 0 : 8;
  int aRowOff = (wr * 64 + mrow) * 80 + abase;
  int bRowOff = (wc * 32 + mrow) * 80 + bbase;

  int rl = t >> 1, half = t & 1;
  const unsigned char* aSrc = A  + (size_t)(i0 + rl) * N_NODES + half * 32;
  const unsigned char* bSrc = Bt + (size_t)(j0 + rl) * N_NODES + half * 32;
  int ldOff = rl * 80 + half * 32;

  unsigned char* aCur = &As[0][0];
  unsigned char* aNxt = &As[1][0];
  unsigned char* bCur = &Bs[0][0];
  unsigned char* bNxt = &Bs[1][0];

  auto compute_chunk = [&](const unsigned char* aC, const unsigned char* bC) {
    v8i afrag[4], bfrag[2];
#pragma unroll
    for (int mi = 0; mi < 4; ++mi) {
      const unsigned char* p = aC + aRowOff + mi * (16 * 80);
      uint2 d0 = *(const uint2*)(p + 0);
      uint2 d1 = *(const uint2*)(p + 16);
      uint2 d2 = *(const uint2*)(p + 32);
      uint2 d3 = *(const uint2*)(p + 48);
      afrag[mi] = (v8i){(int)d0.x, (int)d0.y, (int)d1.x, (int)d1.y,
                        (int)d2.x, (int)d2.y, (int)d3.x, (int)d3.y};
    }
#pragma unroll
    for (int nj = 0; nj < 2; ++nj) {
      const unsigned char* p = bC + bRowOff + nj * (16 * 80);
      uint2 q0 = *(const uint2*)(p + 0);
      uint2 q1 = *(const uint2*)(p + 16);
      uint2 q2 = *(const uint2*)(p + 32);
      uint2 q3 = *(const uint2*)(p + 48);
      bfrag[nj] = (v8i){(int)q0.x, (int)q0.y, (int)q1.x, (int)q1.y,
                        (int)q2.x, (int)q2.y, (int)q3.x, (int)q3.y};
    }
#pragma unroll
    for (int mi = 0; mi < 4; ++mi)
#pragma unroll
      for (int nj = 0; nj < 2; ++nj)
        acc[mi][nj] = __builtin_amdgcn_wmma_i32_16x16x64_iu8(
            false, afrag[mi], false, bfrag[nj], acc[mi][nj], false, false);
  };

  const int NCHUNK = N_NODES / 64;

#if USE_ASYNC_LDS
  async_copy32B(aSrc, aCur + ldOff);
  async_copy32B(bSrc, bCur + ldOff);

#pragma unroll 1
  for (int n = 0; n < NCHUNK - 1; ++n) {
    async_copy32B(aSrc + n * 64 + 64, aNxt + ldOff);
    async_copy32B(bSrc + n * 64 + 64, bNxt + ldOff);
    WAIT_ASYNC(4);
    __syncthreads();
    compute_chunk(aCur, bCur);
    __syncthreads();
    unsigned char* tmp;
    tmp = aCur; aCur = aNxt; aNxt = tmp;
    tmp = bCur; bCur = bNxt; bNxt = tmp;
  }
  WAIT_ASYNC(0);
  __syncthreads();
  compute_chunk(aCur, bCur);
#else
#pragma unroll 1
  for (int n = 0; n < NCHUNK; ++n) {
    int k0 = n * 64;
    uint4 av0 = *(const uint4*)(aSrc + k0);
    uint4 av1 = *(const uint4*)(aSrc + k0 + 16);
    uint4 bv0 = *(const uint4*)(bSrc + k0);
    uint4 bv1 = *(const uint4*)(bSrc + k0 + 16);
    __syncthreads();
    *(uint4*)(aCur + ldOff)      = av0;
    *(uint4*)(aCur + ldOff + 16) = av1;
    *(uint4*)(bCur + ldOff)      = bv0;
    *(uint4*)(bCur + ldOff + 16) = bv1;
    __syncthreads();
    compute_chunk(aCur, bCur);
    unsigned char* tmp;
    tmp = aCur; aCur = aNxt; aNxt = tmp;
    tmp = bCur; bCur = bNxt; bNxt = tmp;
  }
#endif

  int colLocal = lane & 15;
  int rowAdd   = (lane < 16) ? 0 : 8;
#pragma unroll
  for (int mi = 0; mi < 4; ++mi)
#pragma unroll
    for (int nj = 0; nj < 2; ++nj) {
      int jg    = j0 + wc * 32 + nj * 16 + colLocal;
      int ibase = i0 + wr * 64 + mi * 16 + rowAdd;
      v8i a = acc[mi][nj];
#pragma unroll
      for (int v = 0; v < 8; ++v) {
        out_g[(size_t)(ibase + v) * KSEL + jg] = (a[v] > 0) ? 1.0f : 0.0f;
      }
    }
}

__global__ __launch_bounds__(256) void norm_kernel(float* __restrict__ out_g) {
  __shared__ float red[8];
  int i = blockIdx.x;
  float* row = out_g + (size_t)i * KSEL;
  int t = threadIdx.x;
  float s = 0.f;
  for (int j = t; j < KSEL; j += 256) s += row[j];
  for (int off = 16; off > 0; off >>= 1) s += __shfl_xor(s, off, 32);
  if ((t & 31) == 0) red[t >> 5] = s;
  __syncthreads();
  float total = red[0] + red[1] + red[2] + red[3] + red[4] + red[5] + red[6] + red[7];
  for (int j = t; j < KSEL; j += 256) row[j] = row[j] / total;
}

extern "C" void kernel_launch(void* const* d_in, const int* in_sizes, int n_in,
                              void* d_out, int out_size, void* d_ws, size_t ws_size,
                              hipStream_t stream) {
  const float* g = (const float*)d_in[0];
  const float* h = (const float*)d_in[1];
  const float* W = (const float*)d_in[2];
  const float* b = (const float*)d_in[3];

  float* out_g   = (float*)d_out;
  float* out_h   = out_g + (size_t)KSEL * KSEL;
  float* out_idx = out_h + (size_t)KSEL * DFEAT;

  char* ws = (char*)d_ws;
  int*           idx_i32 = (int*)  (ws + 0);
  float*         vals    = (float*)(ws + 16384);
  int*           rank    = (int*)  (ws + 32768);
  float*         scores  = (float*)(ws + 65536);
  unsigned char* Abuf    = (unsigned char*)(ws + 98304);
  unsigned char* Btbuf   = Abuf + (size_t)KSEL * N_NODES;

  scores_kernel<<<N_NODES / 8, 256, 0, stream>>>(h, W, b, scores);
  topk_kernel<<<1, 1024, 0, stream>>>(scores, idx_i32, vals, rank, out_idx);
  newh_kernel<<<KSEL, 256, 0, stream>>>(h, idx_i32, vals, out_h);
  pack_kernel<<<(N_NODES / 64) * (N_NODES / 64), 256, 0, stream>>>(g, rank, Abuf, Btbuf);
  dim3 ggrid(KSEL / 128, KSEL / 128);
  gemm2hop_kernel<<<ggrid, 256, 0, stream>>>(Abuf, Btbuf, out_g);
  norm_kernel<<<KSEL, 256, 0, stream>>>(out_g);
}
